// Decoder_9234179686941
// MI455X (gfx1250) — compile-verified, matrix operand order checked by reading, not run on hardware
//
#include <hip/hip_runtime.h>
#include <hip/hip_bf16.h>
#include <stdint.h>

typedef __attribute__((ext_vector_type(16))) _Float16 v16h;
typedef __attribute__((ext_vector_type(8)))  float    v8f;
typedef __attribute__((ext_vector_type(4)))  unsigned int u32x4;
typedef __attribute__((ext_vector_type(8)))  int          i32x8;
typedef __attribute__((ext_vector_type(4)))  int          i32x4;

#define NUM_B   128
#define NUM_G   100
#define NUM_KV  10
#define EN      128
#define DM      256
#define NH      8
#define HD      32
#define NKEYS   (NUM_KV * NUM_G)   // 1000
#define NEG_INF_F (-1e15f)

#if defined(__has_builtin)
#if __has_builtin(__builtin_amdgcn_tensor_load_to_lds) && __has_builtin(__builtin_amdgcn_s_wait_tensorcnt)
#define USE_TDM 1
#endif
#endif

// ---------------------------------------------------------------------------
// A-fragment loader: 16x32 f16 tile from LDS, row-major with given stride.
// ISA layout: lanes 0-15 hold row M=lane, K in {0..7, 16..23}; lanes 16-31 hold
// row M=lane-16, K in {8..15, 24..31}.  (Same layout serves B = W^T fragments,
// which we pre-pack in memory so each lane reads 16 contiguous halves.)
// ---------------------------------------------------------------------------
__device__ inline v16h load_a_frag(const _Float16* base, int lane, int kt, int stride) {
    int row = lane & 15;
    int kb0 = (lane >> 4) << 3;                 // 0 or 8
    const _Float16* p = base + row * stride + kt * 32 + kb0;
    v16h a;
#pragma unroll
    for (int e = 0; e < 8; ++e)  a[e] = p[e];       // K = kb0 + e
#pragma unroll
    for (int e = 8; e < 16; ++e) a[e] = p[e + 8];   // K = kb0 + 16 + (e-8)
    return a;
}

#ifdef USE_TDM
// ---------------------------------------------------------------------------
// Tensor Data Mover: async 2D tile load (rows x 256 f32) global -> LDS.
// D# per CDNA5 ISA 8.3-8.5: group0 = {flags, lds_addr, global_addr, type=2},
// group1 = {mask/data_size, dims/tile dims/strides}.  OOB rows zero-fill.
// ---------------------------------------------------------------------------
__device__ inline void tdm_load_tile(const float* gsrc, unsigned lds_off,
                                     unsigned rows_valid) {
    unsigned long long ga = (unsigned long long)(uintptr_t)gsrc;
    u32x4 g0 = (u32x4)0u;
    g0[0] = 1u;                                        // count=1, user mode
    g0[1] = lds_off;                                   // LDS byte address
    g0[2] = (unsigned)(ga & 0xFFFFFFFFu);              // global_addr lo
    g0[3] = (unsigned)((ga >> 32) & 0x01FFFFFFu) | (2u << 30);  // hi | type=2
    i32x8 g1 = (i32x8)0;
    g1[0] = (int)(2u << 16);                           // data_size = 4 bytes
    g1[1] = (int)(256u << 16);                         // tensor_dim0 = 256 (lo16)
    g1[2] = (int)(rows_valid << 16);                   // dim0 hi=0 | tensor_dim1 lo
    g1[3] = (int)(256u << 16);                         // dim1 hi=0 | tile_dim0 = 256
    g1[4] = (int)16;                                   // tile_dim1 = 16, tile_dim2 = 0
    g1[5] = (int)256;                                  // tensor_dim0_stride = 256
    g1[6] = 0;
    g1[7] = 0;
    i32x4 gz = (i32x4)0;
#if __clang_major__ >= 23
    __builtin_amdgcn_tensor_load_to_lds(g0, g1, gz, gz, (i32x8)0, 0);
#else
    __builtin_amdgcn_tensor_load_to_lds(g0, g1, gz, gz, 0);
#endif
}
#endif

// ---------------------------------------------------------------------------
// Kernel 1: pack Wnv / Wk / Wv / Wlg into f16 WMMA-B-fragment order.
// Packed linear index == ((ntile*nKt + kt)*32 + lane)*16 + e.
// ---------------------------------------------------------------------------
__global__ __launch_bounds__(256) void pack_weights(
    const float* __restrict__ Wk, const float* __restrict__ Wv,
    const float* __restrict__ Wlg, const float* __restrict__ Wnv,
    _Float16* __restrict__ wkP, _Float16* __restrict__ wvP,
    _Float16* __restrict__ wlgP, _Float16* __restrict__ wnvP) {
    int idx  = blockIdx.x * blockDim.x + threadIdx.x;   // 0..65535
    int e    = idx & 15;
    int lane = (idx >> 4) & 31;
    int kb0  = (lane >> 4) << 3;
    int ke   = kb0 + (e < 8 ? e : e + 8);
    int nloc = lane & 15;
    {   // 256x256 weights: nKt = 8, 16 ntiles
        int kt    = (idx >> 9) & 7;
        int ntile = idx >> 12;
        int nout  = ntile * 16 + nloc;
        int kin   = kt * 32 + ke;
        wkP[idx]  = (_Float16)Wk[nout * DM + kin];
        wvP[idx]  = (_Float16)Wv[nout * DM + kin];
        wlgP[idx] = (_Float16)Wlg[nout * DM + kin];
    }
    if (idx < 32768) {   // Wnv [256,129]: use cols 1..128 (col 0 = mm coeff)
        int kt    = (idx >> 9) & 3;
        int ntile = idx >> 11;
        int nout  = ntile * 16 + nloc;
        int kin   = kt * 32 + ke + 1;
        wnvP[idx] = (_Float16)Wnv[nout * 129 + kin];
    }
}

// ---------------------------------------------------------------------------
// Kernel 2: Q = [h_g|fleet|act|h[:,0,:]|last] @ Wq^T + bq   (tiny, VALU)
// ---------------------------------------------------------------------------
__global__ __launch_bounds__(256) void compute_q(
    const float* __restrict__ h, const float* __restrict__ h_g,
    const float* __restrict__ fleet, const float* __restrict__ act,
    const float* __restrict__ last, const float* __restrict__ Wq,
    const float* __restrict__ bq, float* __restrict__ qws) {
    int b = blockIdx.x, o = threadIdx.x;
    const float* segs[5] = { h_g + b * EN, fleet + b * EN, act + b * EN,
                             h + (size_t)b * NUM_G * EN, last + b * EN };
    const float* wrow = Wq + (size_t)o * (5 * EN);
    float acc = bq[o];
#pragma unroll 1
    for (int s = 0; s < 5; ++s) {
        const float* sp = segs[s];
        const float* wp = wrow + s * EN;
        for (int e = 0; e < EN; ++e) acc += sp[e] * wp[e];
    }
    qws[b * DM + o] = acc;
}

// ---------------------------------------------------------------------------
// Kernel 3: fused g_a -> {K->scores, V, K_lg}.  Grid (kv, b), 128 thr = 4 waves.
// All waves cooperate on one 16-row M-tile; each wave owns 4 of 16 N-tiles.
// Precomputed additive tile is double-buffered in LDS and streamed in by the
// Tensor Data Mover, overlapped with the WMMA chains of the previous tile.
// ---------------------------------------------------------------------------
__global__ __launch_bounds__(128) void fused_ga_kvlg(
    const float* __restrict__ h, const float* __restrict__ vehicle,
    const float* __restrict__ precomputed,
    const float* __restrict__ Wveh, const float* __restrict__ bveh,
    const float* __restrict__ Wnv,  const float* __restrict__ bnv,
    const float* __restrict__ bk,   const float* __restrict__ bv,
    const float* __restrict__ blg,
    const _Float16* __restrict__ wnvP, const _Float16* __restrict__ wkP,
    const _Float16* __restrict__ wvP,  const _Float16* __restrict__ wlgP,
    const float* __restrict__ qws,
    float* __restrict__ scores, _Float16* __restrict__ Vws,
    _Float16* __restrict__ Klgws) {

    __shared__ float    veh_sh[EN];
    __shared__ float    q_sh[DM];
    __shared__ float    vproj_sh[DM];
    __shared__ float    mm_sh[16];
    __shared__ _Float16 A_sh[16 * EN];        // pwm tile (block-shared, 4KB)
    __shared__ _Float16 Ga_sh[16 * DM];       // g_a tile f16 (8KB)
    __shared__ float    pre_sh[2][16 * DM];   // TDM double buffer (32KB)
    __shared__ float    Ktmp_sh[4][16 * 32];  // per-wave K slab (8KB)

    const int b    = blockIdx.y;
    const int kk   = blockIdx.x;
    const int tid  = threadIdx.x;
    const int lane = tid & 31;
    const int wave = tid >> 5;

    // ---- block-wide preamble ----------------------------------------------
    veh_sh[tid]     = vehicle[((size_t)b * NUM_KV + kk) * EN + tid];
    q_sh[tid]       = qws[b * DM + tid];
    q_sh[tid + 128] = qws[b * DM + tid + 128];
    if (tid < 16) mm_sh[tid] = 0.f;
    __syncthreads();
    for (int o = tid; o < DM; o += 128) {          // veh_proj = Wveh@veh + bveh
        float acc = bveh[o];
        const float* wp = Wveh + (size_t)o * EN;
        for (int e = 0; e < EN; ++e) acc += wp[e] * veh_sh[e];
        vproj_sh[o] = acc;
    }
    __syncthreads();

    const float* preb = precomputed + (size_t)b * NUM_G * DM;
#ifdef USE_TDM
    if (wave == 0)     // prefetch tile 0's additive slab
        tdm_load_tile(preb, (unsigned)(uintptr_t)&pre_sh[0][0], NUM_G);
#endif

    // 100 g-rows -> 7 M-tiles of 16 rows, processed by the whole block
#pragma unroll 1
    for (int t = 0; t < 7; ++t) {
        const int buf = t & 1;
        // ---- build pwm tile (coalesced) + row sums via LDS atomics ---------
        {
            int row = tid >> 3;                 // 0..15
            int e0  = (tid & 7) * 16;           // 16-elem chunk
            int gi  = t * 16 + row;
            bool valid = gi < NUM_G;
            const float* hrow = h + ((size_t)b * NUM_G + (valid ? gi : 0)) * EN + e0;
            float part = 0.f;
#pragma unroll
            for (int j = 0; j < 16; ++j) {
                float pv = valid ? veh_sh[e0 + j] * hrow[j] : 0.f;
                A_sh[row * EN + e0 + j] = (_Float16)pv;
                part += pv;
            }
            atomicAdd(&mm_sh[row], part);
        }
#ifdef USE_TDM
        if (wave == 0) __builtin_amdgcn_s_wait_tensorcnt(0);
#else
        for (int idx = tid; idx < 16 * DM; idx += 128) {
            int row = idx >> 8, col = idx & 255;
            int gi = t * 16 + row;
            pre_sh[buf][idx] = (gi < NUM_G) ? preb[(size_t)gi * DM + col] : 0.f;
        }
#endif
        __syncthreads();   // pwm + mm + pre[buf] ready for all waves
#ifdef USE_TDM
        if (wave == 0 && t + 1 < 7)             // overlap next tile's DMA
            tdm_load_tile(preb + (size_t)(t + 1) * 16 * DM,
                          (unsigned)(uintptr_t)&pre_sh[buf ^ 1][0],
                          (unsigned)(NUM_G - (t + 1) * 16));
#endif

        // ---- GEMM1: g_a = pwm @ WnvP (K=128), 4 N-tiles per wave -----------
#pragma unroll 1
        for (int nt = wave * 4; nt < wave * 4 + 4; ++nt) {
            v8f acc = {};
#pragma unroll
            for (int kt = 0; kt < 4; ++kt) {
                v16h a  = load_a_frag(A_sh, lane, kt, EN);
                v16h bf = *(const v16h*)(wnvP + (((nt * 4 + kt) * 32) + lane) * 16);
                acc = __builtin_amdgcn_wmma_f32_16x16x32_f16(
                        false, a, false, bf, (short)0, acc, false, false);
            }
            int col  = nt * 16 + (lane & 15);
            int r0   = (lane >> 4) * 8;
            float c0 = bnv[col] + vproj_sh[col];
            float w0 = Wnv[col * 129];           // mm coefficient (col 0)
#pragma unroll
            for (int v = 0; v < 8; ++v) {
                int row = r0 + v;
                float x = acc[v] + c0 + mm_sh[row] * w0
                        + pre_sh[buf][row * DM + col];
                Ga_sh[row * DM + col] = (_Float16)x;
            }
        }
        __syncthreads();   // Ga tile complete (cross-wave producer/consumer)

        // ---- GEMM2 x3: K (-> scores), V, K_lg (K=256), 4 N-tiles per wave --
        const _Float16* Wp[3] = { wkP, wvP, wlgP };
        const float*    Bp[3] = { bk, bv, blg };
        _Float16*       Op[3] = { nullptr, Vws, Klgws };
#pragma unroll 1
        for (int m = 0; m < 3; ++m) {
#pragma unroll 1
            for (int nt = wave * 4; nt < wave * 4 + 4; ++nt) {
                v8f acc = {};
#pragma unroll
                for (int kt = 0; kt < 8; ++kt) {
                    v16h a  = load_a_frag(Ga_sh, lane, kt, DM);
                    v16h bf = *(const v16h*)(Wp[m] + (((nt * 8 + kt) * 32) + lane) * 16);
                    acc = __builtin_amdgcn_wmma_f32_16x16x32_f16(
                            false, a, false, bf, (short)0, acc, false, false);
                }
                int col = nt * 16 + (lane & 15);
                int r0  = (lane >> 4) * 8;
                float bb = Bp[m][col];
                if (m == 0) {
                    float* Ktw = Ktmp_sh[wave];
#pragma unroll
                    for (int v = 0; v < 8; ++v)
                        Ktw[(r0 + v) * 32 + (nt & 1) * 16 + (lane & 15)] = acc[v] + bb;
                    if (nt & 1) {                 // head (2 N-tiles) complete
                        int hh = nt >> 1;
                        if (lane < 16) {
                            int row = lane, gi = t * 16 + row;
                            float s = 0.f;
#pragma unroll
                            for (int c = 0; c < HD; ++c)
                                s += Ktw[row * 32 + c] * q_sh[hh * HD + c];
                            s *= 0.17677669529663687f;   // 1/sqrt(32)
                            if (gi < NUM_G)
                                scores[((size_t)b * NH + hh) * NKEYS
                                       + kk * NUM_G + gi] = s;
                        }
                    }
                } else {
                    _Float16* dst = Op[m];
#pragma unroll
                    for (int v = 0; v < 8; ++v) {
                        int row = r0 + v, gi = t * 16 + row;
                        if (gi < NUM_G)
                            dst[((size_t)b * NKEYS + kk * NUM_G + gi) * DM + col]
                                = (_Float16)(acc[v] + bb);
                    }
                }
            }
        }
        if (tid < 16) mm_sh[tid] = 0.f;          // reset row sums for next tile
        __syncthreads();   // protect A_sh/Ga_sh/mm_sh before next iteration
    }
}

// ---------------------------------------------------------------------------
// Kernel 4: masked softmax per head, u = attn@V, Wo, tanh logits vs K_lg,
// block argmax + prob.  One block per batch row.
// ---------------------------------------------------------------------------
__global__ __launch_bounds__(256) void attn_finalize(
    const float* __restrict__ scores, const _Float16* __restrict__ Vws,
    const _Float16* __restrict__ Klgws, const int* __restrict__ mask,
    const float* __restrict__ Wo, const float* __restrict__ bo,
    float* __restrict__ out) {

    __shared__ float sc_sh[NH * NKEYS];
    __shared__ float u_sh[DM];
    __shared__ float u2_sh[DM];
    __shared__ float logit_sh[NKEYS];
    __shared__ float redf[16];
    __shared__ int   redi[16];

    const int b = blockIdx.x, tid = threadIdx.x, lane = tid & 31, w = tid >> 5;
    const int* mrow = mask + b * NKEYS;

    for (int j = tid; j < NH * NKEYS; j += 256) {
        int n = j % NKEYS;
        float s = scores[(size_t)b * NH * NKEYS + j];
        sc_sh[j] = (mrow[n] == 0) ? NEG_INF_F : s;
    }
    __syncthreads();

    // one wave per head
    {
        int hh = w;
        float m = -3.4e38f;
        for (int n = lane; n < NKEYS; n += 32) m = fmaxf(m, sc_sh[hh * NKEYS + n]);
#pragma unroll
        for (int o = 16; o > 0; o >>= 1) m = fmaxf(m, __shfl_xor(m, o, 32));
        float s = 0.f;
        for (int n = lane; n < NKEYS; n += 32) {
            float e = expf(sc_sh[hh * NKEYS + n] - m);
            sc_sh[hh * NKEYS + n] = e;
            s += e;
        }
#pragma unroll
        for (int o = 16; o > 0; o >>= 1) s += __shfl_xor(s, o, 32);
        float rinv = 1.f / s;
        const _Float16* Vb = Vws + (size_t)b * NKEYS * DM + hh * HD + lane;
        float ud = 0.f;
        for (int n = 0; n < NKEYS; ++n)
            ud += sc_sh[hh * NKEYS + n] * (float)Vb[(size_t)n * DM];
        u_sh[hh * HD + lane] = ud * rinv;
    }
    __syncthreads();

    {   // u2 = Wo @ u + bo
        float acc = bo[tid];
        const float* wp = Wo + (size_t)tid * DM;
        for (int d = 0; d < DM; ++d) acc += wp[d] * u_sh[d];
        u2_sh[tid] = acc;
    }
    __syncthreads();

    const _Float16* Kb = Klgws + (size_t)b * NKEYS * DM;
    for (int n = tid; n < NKEYS; n += 256) {
        float dot = 0.f;
        const _Float16* kr = Kb + (size_t)n * DM;
        for (int d = 0; d < DM; ++d) dot += u2_sh[d] * (float)kr[d];
        float lg = 10.f * tanhf(dot * 0.0625f);        // /sqrt(256)
        logit_sh[n] = (mrow[n] == 0) ? NEG_INF_F : lg;
    }
    __syncthreads();

    // block argmax (first index wins ties, matching jnp.argmax)
    float bm = -3.4e38f; int bi = 0;
    for (int n = tid; n < NKEYS; n += 256) {
        float l = logit_sh[n];
        if (l > bm) { bm = l; bi = n; }
    }
#pragma unroll
    for (int o = 16; o > 0; o >>= 1) {
        float om = __shfl_xor(bm, o, 32);
        int   oi = __shfl_xor(bi, o, 32);
        if (om > bm || (om == bm && oi < bi)) { bm = om; bi = oi; }
    }
    if (lane == 0) { redf[w] = bm; redi[w] = bi; }
    __syncthreads();
    if (tid == 0) {
        float M = redf[0]; int I = redi[0];
        for (int q = 1; q < 8; ++q)
            if (redf[q] > M || (redf[q] == M && redi[q] < I)) { M = redf[q]; I = redi[q]; }
        redf[8] = M; redi[8] = I;
    }
    __syncthreads();
    float M = redf[8]; int I = redi[8];

    float se = 0.f;
    for (int n = tid; n < NKEYS; n += 256) se += expf(logit_sh[n] - M);
#pragma unroll
    for (int o = 16; o > 0; o >>= 1) se += __shfl_xor(se, o, 32);
    if (lane == 0) redf[w] = se;
    __syncthreads();
    if (tid == 0) {
        float S = 0.f;
        for (int q = 0; q < 8; ++q) S += redf[q];
        out[b * 2 + 0]       = (float)(I / NUM_G);   // kk
        out[b * 2 + 1]       = (float)(I % NUM_G);   // ii
        out[NUM_B * 2 + b]   = 1.f / S;              // prob of argmax
    }
}

// ---------------------------------------------------------------------------
extern "C" void kernel_launch(void* const* d_in, const int* in_sizes, int n_in,
                              void* d_out, int out_size, void* d_ws, size_t ws_size,
                              hipStream_t stream) {
    const float* h      = (const float*)d_in[0];
    const float* h_g    = (const float*)d_in[1];
    const float* veh    = (const float*)d_in[2];
    const float* fleet  = (const float*)d_in[3];
    const float* act    = (const float*)d_in[4];
    const float* last   = (const float*)d_in[5];
    const float* pre    = (const float*)d_in[6];
    const int*   mask   = (const int*)d_in[7];
    const float* Wq  = (const float*)d_in[8];
    const float* bq  = (const float*)d_in[9];
    const float* Wveh= (const float*)d_in[10];
    const float* bveh= (const float*)d_in[11];
    const float* Wnv = (const float*)d_in[12];
    const float* bnv = (const float*)d_in[13];
    const float* Wk  = (const float*)d_in[14];
    const float* bk  = (const float*)d_in[15];
    const float* Wv  = (const float*)d_in[16];
    const float* bv  = (const float*)d_in[17];
    const float* Wlg = (const float*)d_in[18];
    const float* blg = (const float*)d_in[19];
    const float* Wo  = (const float*)d_in[20];
    const float* bo  = (const float*)d_in[21];

    char* ws = (char*)d_ws;
    size_t off = 0;
    auto take = [&](size_t bytes) -> void* {
        void* p = ws + off;
        off += (bytes + 255) & ~(size_t)255;
        return p;
    };
    float*    qws    = (float*)take((size_t)NUM_B * DM * 4);
    float*    scores = (float*)take((size_t)NUM_B * NH * NKEYS * 4);
    _Float16* Vws    = (_Float16*)take((size_t)NUM_B * NKEYS * DM * 2);
    _Float16* Klgws  = (_Float16*)take((size_t)NUM_B * NKEYS * DM * 2);
    _Float16* wkP    = (_Float16*)take((size_t)65536 * 2);
    _Float16* wvP    = (_Float16*)take((size_t)65536 * 2);
    _Float16* wlgP   = (_Float16*)take((size_t)65536 * 2);
    _Float16* wnvP   = (_Float16*)take((size_t)32768 * 2);

    pack_weights<<<256, 256, 0, stream>>>(Wk, Wv, Wlg, Wnv, wkP, wvP, wlgP, wnvP);
    compute_q<<<NUM_B, 256, 0, stream>>>(h, h_g, fleet, act, last, Wq, bq, qws);
    fused_ga_kvlg<<<dim3(NUM_KV, NUM_B), 128, 0, stream>>>(
        h, veh, pre, Wveh, bveh, Wnv, bnv, bk, bv, blg,
        wnvP, wkP, wvP, wlgP, qws, scores, Vws, Klgws);
    attn_finalize<<<NUM_B, 256, 0, stream>>>(scores, Vws, Klgws, mask, Wo, bo,
                                             (float*)d_out);
}
